// MixtureOfExperts_57045755625494
// MI455X (gfx1250) — compile-verified, matrix operand order checked by reading, not run on hardware
//
#include <hip/hip_runtime.h>
#include <hip/hip_bf16.h>
#include <math.h>

#define T_TOK 4096   // B*S
#define HDIM  768
#define IDIM  3072
#define NEXP  8
#define KSEL  2
#define ICHK  128    // I-chunk width

typedef __bf16 bf16;
typedef __attribute__((ext_vector_type(16))) __bf16 v16bf;
typedef __attribute__((ext_vector_type(8)))  float  v8f;
typedef __attribute__((ext_vector_type(4)))  unsigned int u32x4;

union FragB16 { v16bf v; u32x4 q[2]; };

template <typename P>
__device__ inline v16bf load_frag(const P* p0, const P* p1) {
    FragB16 f;
    f.q[0] = *(const u32x4*)p0;
    f.q[1] = *(const u32x4*)p1;
    return f.v;
}

// ---------------- init: zero counters / aux accumulators ----------------
__global__ void moe_init_kernel(int* ecount, float* aux) {
    int i = threadIdx.x;
    if (i < NEXP) ecount[i] = 0;
    if (i < 16)   aux[i]    = 0.0f;
}

// ---------------- fp32 -> bf16 conversion (grid-stride) ----------------
__global__ void moe_cvt_bf16_kernel(const float* __restrict__ in,
                                    bf16* __restrict__ out, int n) {
    int stride = gridDim.x * blockDim.x;
    for (int i = blockIdx.x * blockDim.x + threadIdx.x; i < n; i += stride)
        out[i] = (bf16)in[i];
}

// ---------------- router: logits, softmax, top-2, capacity, aux --------
__global__ __launch_bounds__(128)
void moe_router_kernel(const float* __restrict__ x,
                       const float* __restrict__ rw,
                       const float* __restrict__ cw,
                       const float* __restrict__ cb,
                       int* __restrict__ ecount,
                       int* __restrict__ perm_tok,
                       float* __restrict__ perm_wt,
                       float* __restrict__ aux) {
    const int t    = blockIdx.x * 4 + (threadIdx.x >> 5);
    const int lane = threadIdx.x & 31;
    if (t >= T_TOK) return;

    float acc[9];
    #pragma unroll
    for (int i = 0; i < 9; ++i) acc[i] = 0.0f;

    const float* xp = x + (size_t)t * HDIM;
    for (int h = lane; h < HDIM; h += 32) {
        float xv = xp[h];
        #pragma unroll
        for (int e = 0; e < NEXP; ++e) acc[e] += xv * rw[e * HDIM + h];
        acc[8] += xv * cw[h];
    }
    #pragma unroll
    for (int i = 0; i < 9; ++i)
        for (int off = 16; off > 0; off >>= 1)
            acc[i] += __shfl_xor(acc[i], off, 32);

    if (lane == 0) {
        float mx = acc[0];
        #pragma unroll
        for (int e = 1; e < NEXP; ++e) mx = fmaxf(mx, acc[e]);
        float pe[NEXP], s = 0.0f;
        #pragma unroll
        for (int e = 0; e < NEXP; ++e) { pe[e] = expf(acc[e] - mx); s += pe[e]; }
        float inv = 1.0f / s;
        #pragma unroll
        for (int e = 0; e < NEXP; ++e) pe[e] *= inv;

        // router_z_loss term: logsumexp over the probabilities
        float zs = 0.0f;
        #pragma unroll
        for (int e = 0; e < NEXP; ++e) zs += expf(pe[e]);
        float z = logf(zs);

        float cap = 1.0f / (1.0f + expf(-(acc[8] + cb[0])));

        // top-2 (first-occurrence tie break like lax.top_k)
        int i1 = 0;
        #pragma unroll
        for (int e = 1; e < NEXP; ++e) if (pe[e] > pe[i1]) i1 = e;
        int i2 = (i1 == 0) ? 1 : 0;
        #pragma unroll
        for (int e = 0; e < NEXP; ++e)
            if (e != i1 && pe[e] > pe[i2]) i2 = e;

        // softmax over the two selected probabilities, scale by capacity
        float e1 = expf(pe[i1]), e2 = expf(pe[i2]);
        float den = 1.0f / (e1 + e2);
        float w1 = e1 * den * cap;
        float w2 = e2 * den * cap;

        int s1 = atomicAdd(&ecount[i1], 1);
        perm_tok[i1 * T_TOK + s1] = t * 2 + 0;
        perm_wt [i1 * T_TOK + s1] = w1;
        int s2 = atomicAdd(&ecount[i2], 1);
        perm_tok[i2 * T_TOK + s2] = t * 2 + 1;
        perm_wt [i2 * T_TOK + s2] = w2;

        #pragma unroll
        for (int e = 0; e < NEXP; ++e) atomicAdd(&aux[e], pe[e]);
        atomicAdd(&aux[8], z);
        atomicAdd(&aux[9], cap);
    }
}

// ---------------- fused SwiGLU expert GEMM (bf16 WMMA) -----------------
// Block: 4 waves (128 threads). Grid: (NEXP, T_TOK/16).
// Each block: one expert, one 16-token tile from the compacted list.
__global__ __launch_bounds__(128)
void moe_expert_gemm_kernel(const bf16* __restrict__ xb,
                            const bf16* __restrict__ gb,
                            const bf16* __restrict__ ub,
                            const bf16* __restrict__ db,
                            const int* __restrict__ perm_tok,
                            const float* __restrict__ perm_wt,
                            const int* __restrict__ ecount,
                            float* __restrict__ ybuf) {
    const int e    = blockIdx.x;
    const int rows = ecount[e];
    const int row0 = blockIdx.y * 16;
    if (row0 >= rows) return;

    __shared__ __align__(16) bf16 xs[16 * HDIM];   // 24 KB
    __shared__ __align__(16) bf16 hs[16 * ICHK];   //  4 KB
    __shared__ int   tks[16];
    __shared__ float wts[16];

    const int tid   = threadIdx.x;
    const int wid   = tid >> 5;
    const int lane  = tid & 31;
    const int nlo   = lane & 15;        // row(M) for A frags / row(N) for B frags
    const int hiSel = (lane >> 4) & 1;  // K half selector per ISA layout

    if (tid < 16) {
        int r  = row0 + tid;
        int rc = (r < rows) ? r : (rows - 1);
        tks[tid] = perm_tok[e * T_TOK + rc];
        wts[tid] = perm_wt [e * T_TOK + rc];
    }
    __syncthreads();

    // stage x tile into LDS (16B chunks, 96 chunks/row)
    for (int c = tid; c < 16 * 96; c += 128) {
        int j = c / 96, off = (c % 96) * 8;
        int t = tks[j] >> 1;
        *(u32x4*)&xs[j * HDIM + off] = *(const u32x4*)&xb[(size_t)t * HDIM + off];
    }
    __syncthreads();

    const v8f vzero = {0.f, 0.f, 0.f, 0.f, 0.f, 0.f, 0.f, 0.f};
    v8f yacc[12];
    #pragma unroll
    for (int j = 0; j < 12; ++j) yacc[j] = vzero;

    for (int i0 = 0; i0 < IDIM; i0 += ICHK) {
        // Phase 1: gate/up tiles for this chunk (wave owns 2 of 8 i-tiles)
        for (int itl = 0; itl < 2; ++itl) {
            const int ib = i0 + (wid * 2 + itl) * 16;
            v8f gacc = vzero, uacc = vzero;
            const bf16* gwp = gb + ((size_t)e * IDIM + ib + nlo) * HDIM;
            const bf16* uwp = ub + ((size_t)e * IDIM + ib + nlo) * HDIM;
            for (int k0 = 0; k0 < HDIM; k0 += 32) {
                const bf16* ap = &xs[nlo * HDIM + k0 + hiSel * 8];
                v16bf a  = load_frag(ap, ap + 16);
                v16bf bg = load_frag(gwp + k0 + hiSel * 8, gwp + k0 + 16 + hiSel * 8);
                v16bf bu = load_frag(uwp + k0 + hiSel * 8, uwp + k0 + 16 + hiSel * 8);
                gacc = __builtin_amdgcn_wmma_f32_16x16x32_bf16(
                           false, a, false, bg, (short)0, gacc, false, false);
                uacc = __builtin_amdgcn_wmma_f32_16x16x32_bf16(
                           false, a, false, bu, (short)0, uacc, false, false);
            }
            // silu(gate)*up -> bf16 hid chunk in LDS
            #pragma unroll
            for (int r = 0; r < 8; ++r) {
                float g = gacc[r], u = uacc[r];
                float hval = (g / (1.0f + __expf(-g))) * u;
                int m = r + hiSel * 8;
                hs[m * ICHK + (wid * 2 + itl) * 16 + nlo] = (bf16)hval;
            }
        }
        __syncthreads();

        // Phase 2: yacc += hid[16,ICHK] @ down[e][:, i-chunk]^T
        for (int k2 = 0; k2 < ICHK; k2 += 32) {
            const bf16* ap = &hs[nlo * ICHK + k2 + hiSel * 8];
            v16bf a2 = load_frag(ap, ap + 16);
            #pragma unroll
            for (int j = 0; j < 12; ++j) {
                const int hb = (wid * 12 + j) * 16;
                const bf16* dwp = db + ((size_t)e * HDIM + hb + nlo) * IDIM
                                     + i0 + k2 + hiSel * 8;
                v16bf b2 = load_frag(dwp, dwp + 16);
                yacc[j] = __builtin_amdgcn_wmma_f32_16x16x32_bf16(
                              false, a2, false, b2, (short)0, yacc[j], false, false);
            }
        }
        __syncthreads();  // hs reused next chunk
    }

    // epilogue: scale by combine weight, scatter to per-(t,k) slot
    #pragma unroll
    for (int j = 0; j < 12; ++j) {
        const int hb = (wid * 12 + j) * 16;
        #pragma unroll
        for (int r = 0; r < 8; ++r) {
            int m = r + hiSel * 8;
            if (row0 + m < rows) {
                int tk = tks[m];
                ybuf[(size_t)tk * HDIM + hb + nlo] = yacc[j][r] * wts[m];
            }
        }
    }
}

// ---------------- combine the K=2 expert contributions ------------------
__global__ void moe_combine_kernel(const float* __restrict__ ybuf,
                                   float* __restrict__ out) {
    int i = blockIdx.x * 256 + threadIdx.x;
    if (i >= T_TOK * HDIM) return;
    int t = i / HDIM, h = i % HDIM;
    out[i] = ybuf[(size_t)(2 * t) * HDIM + h]
           + ybuf[(size_t)(2 * t + 1) * HDIM + h];
}

// ---------------- finalize the 4 scalar aux losses ----------------------
__global__ void moe_finalize_kernel(const int* __restrict__ ecount,
                                    const float* __restrict__ aux,
                                    float* __restrict__ losses) {
    if (threadIdx.x != 0 || blockIdx.x != 0) return;
    const float total = (float)(T_TOK * KSEL);
    const float mean  = total / (float)NEXP;
    float var = 0.0f;
    for (int e = 0; e < NEXP; ++e) {
        float d = (float)ecount[e] - mean;
        var += d * d;
    }
    var /= (float)(NEXP - 1);
    losses[0] = var / (mean * mean);                 // load_balance_loss
    losses[1] = aux[8] / (float)T_TOK;               // router_z_loss
    float div = 0.0f;
    for (int e = 0; e < NEXP; ++e) {
        float p = aux[e] / (float)T_TOK;
        div -= p * logf(p + 1e-8f);
    }
    losses[2] = div;                                 // diversity_loss
    float cm = aux[9] / (float)T_TOK - 0.6f;
    losses[3] = cm * cm;                             // capacity_loss
}

// ------------------------------------------------------------------------
extern "C" void kernel_launch(void* const* d_in, const int* in_sizes, int n_in,
                              void* d_out, int out_size, void* d_ws, size_t ws_size,
                              hipStream_t stream) {
    const float* x   = (const float*)d_in[0];
    const float* rw  = (const float*)d_in[1];
    const float* cw  = (const float*)d_in[2];
    const float* cb  = (const float*)d_in[3];
    const float* gw  = (const float*)d_in[4];
    const float* uw  = (const float*)d_in[5];
    const float* dw  = (const float*)d_in[6];
    float* out = (float*)d_out;

    // workspace layout (256B aligned)
    char* ws = (char*)d_ws;
    size_t off = 0;
    auto alloc = [&](size_t bytes) {
        char* p = ws + off;
        off += (bytes + 255) & ~(size_t)255;
        return p;
    };
    const size_t NX = (size_t)T_TOK * HDIM;
    const size_t NW = (size_t)NEXP * IDIM * HDIM;
    bf16*  xb       = (bf16*)alloc(NX * sizeof(bf16));
    bf16*  gb       = (bf16*)alloc(NW * sizeof(bf16));
    bf16*  ub       = (bf16*)alloc(NW * sizeof(bf16));
    bf16*  db       = (bf16*)alloc(NW * sizeof(bf16));
    int*   perm_tok = (int*)  alloc((size_t)NEXP * T_TOK * sizeof(int));
    float* perm_wt  = (float*)alloc((size_t)NEXP * T_TOK * sizeof(float));
    int*   ecount   = (int*)  alloc(NEXP * sizeof(int));
    float* aux      = (float*)alloc(16 * sizeof(float));
    float* ybuf     = (float*)alloc((size_t)T_TOK * KSEL * HDIM * sizeof(float));
    (void)ws_size; (void)in_sizes; (void)n_in; (void)out_size;

    moe_init_kernel<<<1, 32, 0, stream>>>(ecount, aux);

    moe_cvt_bf16_kernel<<<1024, 256, 0, stream>>>(x,  xb, (int)NX);
    moe_cvt_bf16_kernel<<<4096, 256, 0, stream>>>(gw, gb, (int)NW);
    moe_cvt_bf16_kernel<<<4096, 256, 0, stream>>>(uw, ub, (int)NW);
    moe_cvt_bf16_kernel<<<4096, 256, 0, stream>>>(dw, db, (int)NW);

    moe_router_kernel<<<T_TOK / 4, 128, 0, stream>>>(
        x, rw, cw, cb, ecount, perm_tok, perm_wt, aux);

    dim3 ggrid(NEXP, T_TOK / 16);
    moe_expert_gemm_kernel<<<ggrid, 128, 0, stream>>>(
        xb, gb, ub, db, perm_tok, perm_wt, ecount, ybuf);

    moe_combine_kernel<<<(T_TOK * HDIM + 255) / 256, 256, 0, stream>>>(ybuf, out);

    moe_finalize_kernel<<<1, 32, 0, stream>>>(ecount, aux, out + (size_t)T_TOK * HDIM);
}
